// myGAE_67791763800385
// MI455X (gfx1250) — matrix operand order repaired and checked by reading, not compiled, not run on hardware
//
#include <hip/hip_runtime.h>
#include <math.h>

#define N_NODES 20000
#define E_EDGES 65536
#define DF 300
#define KBLK 10          // K padded to 320 = 10 x 32

typedef __attribute__((ext_vector_type(16))) __bf16 v16bf;
typedef __attribute__((ext_vector_type(16))) short  v16s;
typedef __attribute__((ext_vector_type(8)))  short  v8s;
typedef __attribute__((ext_vector_type(8)))  float  v8f;

// ---------------- helpers -------------------------------------------------
__device__ __forceinline__ unsigned short f32_to_bf16(float x) {
  unsigned u = __float_as_uint(x);
  u += 0x7FFFu + ((u >> 16) & 1u);          // round-to-nearest-even
  return (unsigned short)(u >> 16);
}
__device__ __forceinline__ float bf16_to_f32(unsigned short h) {
  return __uint_as_float(((unsigned)h) << 16);
}
__device__ __forceinline__ unsigned f2ord(float f) {
  unsigned u = __float_as_uint(f);
  return (u & 0x80000000u) ? ~u : (u | 0x80000000u);
}
__device__ __forceinline__ float ord2f(unsigned u) {
  return (u & 0x80000000u) ? __uint_as_float(u & 0x7FFFFFFFu)
                           : __uint_as_float(~u);
}
// one 16-bf16 fragment = two contiguous b128 loads
__device__ __forceinline__ v16bf frag_load(const unsigned short* base) {
  const v8s* p = (const v8s*)base;
  v8s x0 = p[0];
  v8s x1 = p[1];
  v16s v = __builtin_shufflevector(x0, x1, 0, 1, 2, 3, 4, 5, 6, 7,
                                   8, 9, 10, 11, 12, 13, 14, 15);
  return __builtin_bit_cast(v16bf, v);
}
__device__ __forceinline__ void split8(const float* xv, v8s& hv, v8s& lv) {
#pragma unroll
  for (int j = 0; j < 8; ++j) {
    unsigned short h = f32_to_bf16(xv[j]);
    unsigned short l = f32_to_bf16(xv[j] - bf16_to_f32(h));
    hv[j] = (short)h;
    lv[j] = (short)l;
  }
}

// ---------------- operand converters (fp32 -> swizzled bf16 hi/lo) --------
// A plane layout: [slab(16 rows)][kb(32 k)][lane][16 halves]
//   lane = ((k&15)>>3)<<4 | (row&15);  h = ((k>>4)&1)*8 + (k&7)
__global__ void convA_kernel(const float* __restrict__ A,
                             unsigned short* __restrict__ hi,
                             unsigned short* __restrict__ lo,
                             int M, int Mpad) {
  size_t idx = (size_t)blockIdx.x * blockDim.x + threadIdx.x;
  size_t total = (size_t)Mpad * 40;            // 40 chunks of 8 k per row
  if (idx >= total) return;
  int row = (int)(idx / 40);
  int k0  = (int)(idx % 40) * 8;               // 0..312
  float xv[8];
  if (row < M && k0 + 7 < DF) {
    const float* p = A + (size_t)row * DF + k0;
#pragma unroll
    for (int j = 0; j < 8; ++j) xv[j] = p[j];
  } else if (row < M) {                        // k boundary chunk (296..303)
#pragma unroll
    for (int j = 0; j < 8; ++j) {
      int k = k0 + j;
      int kc = (k < DF) ? k : (DF - 1);        // clamp addr, select value
      float x = A[(size_t)row * DF + kc];
      xv[j] = (k < DF) ? x : 0.f;
    }
  } else {
#pragma unroll
    for (int j = 0; j < 8; ++j) xv[j] = 0.f;
  }
  v8s hv, lv;
  split8(xv, hv, lv);
  int slab = row >> 4;
  int kb   = k0 >> 5;
  int kin  = k0 & 31;                          // 0,8,16,24
  int lane = (((kin & 15) >> 3) << 4) | (row & 15);
  int h0   = (kin >> 4) * 8;
  size_t off = (((size_t)slab * KBLK + kb) * 32 + lane) * 16 + h0;
  *(v8s*)(hi + off) = hv;
  *(v8s*)(lo + off) = lv;
}

// B plane layout: [kb][subtile(16 cols)][lane][16 halves]
//   lane = ((k&15)>>3)<<4 | (col&15);  h = ((k>>4)&1)*8 + (k&7)
__global__ void convB_kernel(const float* __restrict__ W,
                             unsigned short* __restrict__ hi,
                             unsigned short* __restrict__ lo,
                             int Nc, int npad) {
  size_t idx = (size_t)blockIdx.x * blockDim.x + threadIdx.x;
  size_t total = (size_t)40 * npad;
  if (idx >= total) return;
  int c  = (int)(idx % npad);
  int k0 = (int)(idx / npad) * 8;
  float xv[8];
  if (c < Nc && k0 + 7 < DF) {
#pragma unroll
    for (int j = 0; j < 8; ++j) xv[j] = W[(size_t)(k0 + j) * Nc + c];
  } else if (c < Nc) {
#pragma unroll
    for (int j = 0; j < 8; ++j) {
      int k = k0 + j;
      int kc = (k < DF) ? k : (DF - 1);
      float x = W[(size_t)kc * Nc + c];
      xv[j] = (k < DF) ? x : 0.f;
    }
  } else {
#pragma unroll
    for (int j = 0; j < 8; ++j) xv[j] = 0.f;
  }
  v8s hv, lv;
  split8(xv, hv, lv);
  int sub = c >> 4;
  int kb  = k0 >> 5;
  int kin = k0 & 31;
  int lane = (((kin & 15) >> 3) << 4) | (c & 15);
  int h0   = (kin >> 4) * 8;                    // 0 or 8
  size_t off = (((size_t)kb * (npad >> 4) + sub) * 32 + lane) * 16 + h0;
  *(v8s*)(hi + off) = hv;
  *(v8s*)(lo + off) = lv;
}

// ---------------- GEMM: C[M x Nc] = A * W + bias (pre-swizzled operands) --
// 256 threads = 8 waves; 128x64 C tile; wave = 16 rows x 64 cols (4 accs).
// 12 WMMAs per K-step in 4 independent 3-deep chains (hazard-free
// interleave); fragments load directly as 2x global b128; no LDS, no syncs.
__global__ __launch_bounds__(256) void gemm_wmma_kernel(
    const unsigned short* __restrict__ Ahi, const unsigned short* __restrict__ Alo,
    const unsigned short* __restrict__ Bhi, const unsigned short* __restrict__ Blo,
    const float* __restrict__ bias, float* __restrict__ C,
    int M, int Nc, int nsub) {
  const int lane = threadIdx.x & 31;
  const int wave = threadIdx.x >> 5;
  const int slab = blockIdx.x * 8 + wave;   // 16-row slab per wave
  const int sub0 = blockIdx.y * 4;          // 4 x 16-col subtiles per wave

  v8f acc[4];
#pragma unroll
  for (int s = 0; s < 4; ++s) acc[s] = (v8f){0.f,0.f,0.f,0.f,0.f,0.f,0.f,0.f};

#pragma unroll 2
  for (int kb = 0; kb < KBLK; ++kb) {
    size_t aoff = (((size_t)slab * KBLK + kb) * 32 + lane) * 16;
    v16bf a_hi = frag_load(Ahi + aoff);
    v16bf a_lo = frag_load(Alo + aoff);
    v16bf bh[4], bl[4];
#pragma unroll
    for (int s = 0; s < 4; ++s) {
      size_t boff = (((size_t)kb * nsub + sub0 + s) * 32 + lane) * 16;
      bh[s] = frag_load(Bhi + boff);
      bl[s] = frag_load(Blo + boff);
    }
#pragma unroll
    for (int s = 0; s < 4; ++s)
      acc[s] = __builtin_amdgcn_wmma_f32_16x16x32_bf16(false, a_hi, false, bh[s],
                                                       (short)0, acc[s], false, false);
#pragma unroll
    for (int s = 0; s < 4; ++s)
      acc[s] = __builtin_amdgcn_wmma_f32_16x16x32_bf16(false, a_hi, false, bl[s],
                                                       (short)0, acc[s], false, false);
#pragma unroll
    for (int s = 0; s < 4; ++s)
      acc[s] = __builtin_amdgcn_wmma_f32_16x16x32_bf16(false, a_lo, false, bh[s],
                                                       (short)0, acc[s], false, false);
  }

  // epilogue: D layout -> M = vgpr + 8*(lane>=16), N = lane&15
  const int rl = lane & 15;
  const int g  = lane >> 4;
  const int rowBase = blockIdx.x * 128 + wave * 16 + 8 * g;
  const int colBase = blockIdx.y * 64 + rl;
#pragma unroll
  for (int s = 0; s < 4; ++s) {
    int gcol = colBase + s * 16;
    if (gcol < Nc) {
      float bv = bias[gcol];
#pragma unroll
      for (int v = 0; v < 8; ++v) {
        int grow = rowBase + v;
        if (grow < M) C[(size_t)grow * Nc + gcol] = acc[s][v] + bv;
      }
    }
  }
}

// ---------------- elementwise / fill --------------------------------------
__global__ void fill_f32_kernel(float* p, float v, size_t n) {
  size_t i = (size_t)blockIdx.x * blockDim.x + threadIdx.x;
  if (i < n) p[i] = v;
}
__global__ void fill_u32_kernel(unsigned* p, unsigned v, size_t n) {
  size_t i = (size_t)blockIdx.x * blockDim.x + threadIdx.x;
  if (i < n) p[i] = v;
}
__global__ void add2_kernel(const float* a, const float* b, float* o, size_t n) {
  size_t i = (size_t)blockIdx.x * blockDim.x + threadIdx.x;
  if (i < n) o[i] = a[i] + b[i];
}
__global__ void add3_kernel(const float* a, const float* b, const float* c,
                            float* o, size_t n) {
  size_t i = (size_t)blockIdx.x * blockDim.x + threadIdx.x;
  if (i < n) o[i] = a[i] + b[i] + c[i];
}
__global__ void copy_kernel(const float* a, float* o, size_t n) {
  size_t i = (size_t)blockIdx.x * blockDim.x + threadIdx.x;
  if (i < n) o[i] = a[i];
}

// ---------------- row L2 normalize (wave per row) -------------------------
__global__ void normalize_kernel(const float* __restrict__ x,
                                 float* __restrict__ y, int rows) {
  int gw = (int)(((size_t)blockIdx.x * blockDim.x + threadIdx.x) >> 5);
  int lane = threadIdx.x & 31;
  if (gw >= rows) return;
  const float* xr = x + (size_t)gw * DF;
  float s = 0.f;
  for (int i = lane; i < DF; i += 32) { float v = xr[i]; s += v * v; }
#pragma unroll
  for (int o = 16; o > 0; o >>= 1) s += __shfl_xor(s, o, 32);
  float inv = 1.0f / fmaxf(sqrtf(s), 1e-12f);
  float* yr = y + (size_t)gw * DF;
  for (int i = lane; i < DF; i += 32) yr[i] = xr[i] * inv;
}

// ---------------- attention edge kernels ----------------------------------
__global__ void edge_dot_kernel(const float* __restrict__ Q,
                                const float* __restrict__ K,
                                const int* __restrict__ src,
                                const int* __restrict__ dst,
                                float* __restrict__ attn, int E, int heads) {
  int gw = (int)(((size_t)blockIdx.x * blockDim.x + threadIdx.x) >> 5);
  int lane = threadIdx.x & 31;
  if (gw >= E * heads) return;
  int e = gw / heads, h = gw % heads;
  int HD = heads * DF;
  const float* q = Q + (size_t)dst[e] * HD + h * DF;
  const float* k = K + (size_t)src[e] * HD + h * DF;
  float acc = 0.f;
  for (int i = lane; i < DF; i += 32) acc += q[i] * k[i];
#pragma unroll
  for (int o = 16; o > 0; o >>= 1) acc += __shfl_xor(acc, o, 32);
  if (lane == 0) attn[gw] = acc * 0.05773502691896258f;  // 1/sqrt(300)
}

__global__ void seg_max_kernel(const float* __restrict__ attn,
                               const int* __restrict__ dst, unsigned* m_u,
                               int E, int heads) {
  int idx = blockIdx.x * blockDim.x + threadIdx.x;
  if (idx >= E * heads) return;
  int e = idx / heads, h = idx % heads;
  atomicMax(&m_u[dst[e] * heads + h], f2ord(attn[idx]));
}

__global__ void exp_sum_kernel(float* __restrict__ attn,
                               const int* __restrict__ dst,
                               const unsigned* __restrict__ m_u,
                               float* __restrict__ s_arr, int E, int heads) {
  int idx = blockIdx.x * blockDim.x + threadIdx.x;
  if (idx >= E * heads) return;
  int e = idx / heads, h = idx % heads;
  float m = ord2f(m_u[dst[e] * heads + h]);
  float ex = expf(attn[idx] - m);
  attn[idx] = ex;
  atomicAdd(&s_arr[dst[e] * heads + h], ex);
}

__global__ void scatter_kernel(const float* __restrict__ attn,
                               const int* __restrict__ dst,
                               const int* __restrict__ src,
                               const float* __restrict__ s_arr,
                               const float* __restrict__ V,
                               float* __restrict__ aggr, int E, int heads) {
  int gw = (int)(((size_t)blockIdx.x * blockDim.x + threadIdx.x) >> 5);
  int lane = threadIdx.x & 31;
  if (gw >= E * heads) return;
  int e = gw / heads, h = gw % heads;
  int HD = heads * DF;
  float w = attn[gw] / s_arr[dst[e] * heads + h];
  const float* vrow = V + (size_t)src[e] * HD + h * DF;
  float* arow = aggr + (size_t)dst[e] * HD + h * DF;
  for (int i = lane; i < DF; i += 32) atomicAdd(&arow[i], w * vrow[i]);
}

__global__ void combine_kernel(const float* __restrict__ aggr,
                               const float* __restrict__ skip,
                               float* __restrict__ out, int n_dst, int heads) {
  size_t idx = (size_t)blockIdx.x * blockDim.x + threadIdx.x;
  size_t total = (size_t)n_dst * DF;
  if (idx >= total) return;
  int n = (int)(idx / DF), d = (int)(idx % DF);
  float s = 0.f;
  for (int h = 0; h < heads; ++h) s += aggr[(size_t)n * heads * DF + h * DF + d];
  out[idx] = s / (float)heads + skip[idx];
}

// ---------------- loss pieces ---------------------------------------------
__global__ void triplet_kernel(const float* __restrict__ rep,
                               const int* __restrict__ batch, float* acc, int B) {
  int gw = (int)(((size_t)blockIdx.x * blockDim.x + threadIdx.x) >> 5);
  int lane = threadIdx.x & 31;
  if (gw >= B) return;
  const float* a = rep + (size_t)batch[gw * 3 + 0] * DF;
  const float* p = rep + (size_t)batch[gw * 3 + 1] * DF;
  const float* n = rep + (size_t)batch[gw * 3 + 2] * DF;
  float sp = 0.f, sn = 0.f;
  for (int i = lane; i < DF; i += 32) {
    float dp = a[i] - p[i] + 1e-6f;
    float dn = a[i] - n[i] + 1e-6f;
    sp += dp * dp;
    sn += dn * dn;
  }
#pragma unroll
  for (int o = 16; o > 0; o >>= 1) {
    sp += __shfl_xor(sp, o, 32);
    sn += __shfl_xor(sn, o, 32);
  }
  if (lane == 0) atomicAdd(acc, fmaxf(sqrtf(sp) - sqrtf(sn) + 0.01f, 0.0f));
}

__global__ void mse_kernel(const float* __restrict__ x,
                           const float* __restrict__ ref, float* acc, int rows) {
  int idx = blockIdx.x * blockDim.x + threadIdx.x;
  if (idx >= rows * DF) return;
  float d = x[idx] - ref[idx];
  atomicAdd(acc, d * d);
}

__global__ void finalize_kernel(const float* __restrict__ acc, float* out) {
  if (blockIdx.x == 0 && threadIdx.x == 0) {
    float gl = acc[0] / 1024.0f;
    float r1 = acc[1] / (128.0f * 300.0f);
    float r2 = acc[2] / (128.0f * 300.0f);
    out[0] = 0.5f * r1 + 0.5f * r2 + 0.001f * gl;
  }
}

// ---------------- host-side helpers ---------------------------------------
static void conv_A(hipStream_t stream, const float* X, int M,
                   unsigned short* hi, unsigned short* lo) {
  int Mpad = ((M + 127) / 128) * 128;
  size_t total = (size_t)Mpad * 40;
  convA_kernel<<<(unsigned)((total + 255) / 256), 256, 0, stream>>>(X, hi, lo, M, Mpad);
}

static void run_gemm(hipStream_t stream, const unsigned short* Ahi,
                     const unsigned short* Alo, const float* W, const float* bias,
                     unsigned short* Bhi, unsigned short* Blo, float* C,
                     int M, int Nc) {
  int ntile = (Nc + 63) / 64;
  int npad = ntile * 64;
  size_t totalB = (size_t)40 * npad;
  convB_kernel<<<(unsigned)((totalB + 255) / 256), 256, 0, stream>>>(W, Bhi, Blo, Nc, npad);
  dim3 grid((M + 127) / 128, ntile);
  gemm_wmma_kernel<<<grid, 256, 0, stream>>>(Ahi, Alo, Bhi, Blo, bias, C, M, Nc, npad >> 4);
}

static void run_conv(hipStream_t stream, const float* x_src, int n_src,
                     const float* x_dst, int n_dst, const int* edge, int heads,
                     void* const* d_in, int pbase, float* out, float* Qb,
                     float* Kb, float* Vb, float* skip, float* aggr,
                     float* attn, unsigned* m_u, float* s_arr,
                     unsigned short* Ad_hi, unsigned short* Ad_lo,
                     unsigned short* As_hi, unsigned short* As_lo,
                     unsigned short* Bh, unsigned short* Bl) {
  const float* Wq = (const float*)d_in[pbase + 0];
  const float* bq = (const float*)d_in[pbase + 1];
  const float* Wk = (const float*)d_in[pbase + 2];
  const float* bk = (const float*)d_in[pbase + 3];
  const float* Wv = (const float*)d_in[pbase + 4];
  const float* bv = (const float*)d_in[pbase + 5];
  const float* Ws = (const float*)d_in[pbase + 6];
  const float* bs = (const float*)d_in[pbase + 7];
  const int* src = edge;
  const int* dst = edge + E_EDGES;
  const int HD = heads * DF;
  dim3 blk(256);

  conv_A(stream, x_dst, n_dst, Ad_hi, Ad_lo);
  const unsigned short* Sh = Ad_hi;
  const unsigned short* Sl = Ad_lo;
  if (x_src != x_dst) {
    conv_A(stream, x_src, n_src, As_hi, As_lo);
    Sh = As_hi; Sl = As_lo;
  }
  run_gemm(stream, Ad_hi, Ad_lo, Wq, bq, Bh, Bl, Qb, n_dst, HD);
  run_gemm(stream, Sh, Sl, Wk, bk, Bh, Bl, Kb, n_src, HD);
  run_gemm(stream, Sh, Sl, Wv, bv, Bh, Bl, Vb, n_src, HD);
  run_gemm(stream, Ad_hi, Ad_lo, Ws, bs, Bh, Bl, skip, n_dst, DF);

  size_t nh = (size_t)n_dst * heads;
  fill_u32_kernel<<<(unsigned)((nh + 255) / 256), blk, 0, stream>>>(
      m_u, 0x007FFFFFu /* f2ord(-inf) */, nh);
  fill_f32_kernel<<<(unsigned)((nh + 255) / 256), blk, 0, stream>>>(s_arr, 0.f, nh);
  size_t na = (size_t)n_dst * HD;
  fill_f32_kernel<<<(unsigned)((na + 255) / 256), blk, 0, stream>>>(aggr, 0.f, na);

  int EH = E_EDGES * heads;
  edge_dot_kernel<<<(unsigned)(((size_t)EH * 32 + 255) / 256), blk, 0, stream>>>(
      Qb, Kb, src, dst, attn, E_EDGES, heads);
  seg_max_kernel<<<(EH + 255) / 256, blk, 0, stream>>>(attn, dst, m_u, E_EDGES, heads);
  exp_sum_kernel<<<(EH + 255) / 256, blk, 0, stream>>>(attn, dst, m_u, s_arr, E_EDGES, heads);
  scatter_kernel<<<(unsigned)(((size_t)EH * 32 + 255) / 256), blk, 0, stream>>>(
      attn, dst, src, s_arr, Vb, aggr, E_EDGES, heads);
  size_t nt = (size_t)n_dst * DF;
  combine_kernel<<<(unsigned)((nt + 255) / 256), blk, 0, stream>>>(aggr, skip, out, n_dst, heads);
}

// ---------------- entry ----------------------------------------------------
extern "C" void kernel_launch(void* const* d_in, const int* in_sizes, int n_in,
                              void* d_out, int out_size, void* d_ws,
                              size_t ws_size, hipStream_t stream) {
  (void)in_sizes; (void)n_in; (void)out_size; (void)ws_size;
  const float* H0 = (const float*)d_in[0];
  const float* Ha = (const float*)d_in[1];
  // params (insertion order): sa1[2..9] sa2[10..17] mm[18..25] sm[26..33]
  const float* dec1_W = (const float*)d_in[34];
  const float* dec1_b = (const float*)d_in[35];
  const float* dec2_W = (const float*)d_in[36];
  const float* dec2_b = (const float*)d_in[37];
  const int* edge_a  = (const int*)d_in[38];
  const int* edge_b  = (const int*)d_in[39];
  const int* edge_mm = (const int*)d_in[40];
  const int* edge_sm = (const int*)d_in[41];
  const int* batch   = (const int*)d_in[42];

  const int N = N_NODES;
  const size_t ND = (size_t)N * DF;
  float* w = (float*)d_ws;
  size_t off = 0;
  auto take = [&](size_t n) -> float* { float* p = w + off; off += n; return p; };
  float* f_H0s   = take(ND);
  float* f_Has   = take(ND);
  float* f_Hensn = take(ND);
  float* f_xsrc  = take(2 * ND);
  float* f_meta  = take(ND);
  float* f_conv2 = take(ND);
  float* f_rep   = take(ND);
  float* f_tmp   = take(ND);
  float* f_tmpn  = take(ND);
  float* f_Q     = take(2 * ND);
  float* f_K     = take(4 * ND);
  float* f_V     = take(4 * ND);
  float* f_skip  = take(ND);
  float* f_aggr  = take(2 * ND);
  float* f_attn  = take((size_t)E_EDGES * 2);
  unsigned* u_m  = (unsigned*)take((size_t)N * 2);
  float* f_s     = take((size_t)N * 2);
  float* f_acc   = take(8);
  // swizzled bf16 operand planes (ushort counts halved into float units)
  const size_t AD_US = (size_t)20096 * 320;   // dst operand rows pad to 128
  const size_t AS_US = (size_t)40064 * 320;   // src operand rows pad to 128
  const size_t B_US  = (size_t)640 * 320;     // weights: up to 640 cols
  unsigned short* uAd_hi = (unsigned short*)take(AD_US / 2);
  unsigned short* uAd_lo = (unsigned short*)take(AD_US / 2);
  unsigned short* uAs_hi = (unsigned short*)take(AS_US / 2);
  unsigned short* uAs_lo = (unsigned short*)take(AS_US / 2);
  unsigned short* uB_hi  = (unsigned short*)take(B_US / 2);
  unsigned short* uB_lo  = (unsigned short*)take(B_US / 2);

  dim3 blk(256);
  unsigned gND = (unsigned)((ND + 255) / 256);
  unsigned gNW = (unsigned)(((size_t)N * 32 + 255) / 256);

  // Stage 1: two self-attention convs (heads=1)
  run_conv(stream, H0, N, H0, N, edge_a, 1, d_in, 2, f_H0s,
           f_Q, f_K, f_V, f_skip, f_aggr, f_attn, u_m, f_s,
           uAd_hi, uAd_lo, uAs_hi, uAs_lo, uB_hi, uB_lo);
  run_conv(stream, Ha, N, Ha, N, edge_b, 1, d_in, 10, f_Has,
           f_Q, f_K, f_V, f_skip, f_aggr, f_attn, u_m, f_s,
           uAd_hi, uAd_lo, uAs_hi, uAs_lo, uB_hi, uB_lo);

  // H_ens and normalizations
  add2_kernel<<<gND, blk, 0, stream>>>(f_H0s, f_Has, f_tmp, ND);
  normalize_kernel<<<gNW, blk, 0, stream>>>(f_H0s, f_xsrc, N);
  normalize_kernel<<<gNW, blk, 0, stream>>>(f_Has, f_xsrc + ND, N);
  normalize_kernel<<<gNW, blk, 0, stream>>>(f_tmp, f_Hensn, N);

  // Stage 2: meta convs (heads=2)
  run_conv(stream, f_Hensn, N, f_Hensn, N, edge_mm, 2, d_in, 18, f_meta,
           f_Q, f_K, f_V, f_skip, f_aggr, f_attn, u_m, f_s,
           uAd_hi, uAd_lo, uAs_hi, uAs_lo, uB_hi, uB_lo);
  run_conv(stream, f_xsrc, 2 * N, f_Hensn, N, edge_sm, 2, d_in, 26, f_conv2,
           f_Q, f_K, f_V, f_skip, f_aggr, f_attn, u_m, f_s,
           uAd_hi, uAd_lo, uAs_hi, uAs_lo, uB_hi, uB_lo);

  // rep = normalize(meta_mm + meta_sm + H_ensn)
  add3_kernel<<<gND, blk, 0, stream>>>(f_meta, f_conv2, f_Hensn, f_tmp, ND);
  normalize_kernel<<<gNW, blk, 0, stream>>>(f_tmp, f_rep, N);

  // losses
  fill_f32_kernel<<<1, blk, 0, stream>>>(f_acc, 0.f, 8);
  triplet_kernel<<<(1024 * 32 + 255) / 256, blk, 0, stream>>>(f_rep, batch, &f_acc[0], 1024);

  conv_A(stream, f_rep, N, uAd_hi, uAd_lo);
  run_gemm(stream, uAd_hi, uAd_lo, dec1_W, dec1_b, uB_hi, uB_lo, f_tmp, N, DF);
  normalize_kernel<<<gNW, blk, 0, stream>>>(f_tmp, f_tmpn, N);
  mse_kernel<<<(128 * DF + 255) / 256, blk, 0, stream>>>(f_tmpn, H0, &f_acc[1], 128);

  run_gemm(stream, uAd_hi, uAd_lo, dec2_W, dec2_b, uB_hi, uB_lo, f_tmp, N, DF);
  normalize_kernel<<<gNW, blk, 0, stream>>>(f_tmp, f_tmpn, N);
  mse_kernel<<<(128 * DF + 255) / 256, blk, 0, stream>>>(f_tmpn, Ha, &f_acc[2], 128);

  float* out = (float*)d_out;
  finalize_kernel<<<1, 32, 0, stream>>>(f_acc, out);
  copy_kernel<<<gND, blk, 0, stream>>>(f_rep, out + 1, ND);
}
